// WindowAttention_21560735826116
// MI455X (gfx1250) — compile-verified, matrix operand order checked by reading, not run on hardware
//
#include <hip/hip_runtime.h>

typedef __attribute__((ext_vector_type(16))) _Float16 v16h;
typedef __attribute__((ext_vector_type(8)))  _Float16 v8h;
typedef __attribute__((ext_vector_type(4)))  _Float16 v4h;
typedef __attribute__((ext_vector_type(8)))  float    v8f;

#define DIM   192
#define NTOK  64
#define NH    6
#define HD    32
#define NWMSK 1024

#define WMMA_F16(a, b, c) \
  __builtin_amdgcn_wmma_f32_16x16x32_f16(false, (a), false, (b), (short)0, (c), false, false)

// Fragment convention (ISA 7.12.2, 16-bit A 16x32 / B 32x16): lane&15 selects the
// row (A) / column (B), lane>>4 selects the K half-group; elements 0..7 cover
// K = 8h..8h+7, elements 8..15 cover K = 16+8h..16+8h+7.

// row-major f16 matrix (global or LDS), 16B-aligned rows: two 128-bit loads
__device__ __forceinline__ v16h frag16(const _Float16* __restrict__ M, int ld, int r0,
                                       int kk, int lane) {
  const _Float16* p = M + (size_t)(r0 + (lane & 15)) * ld + kk + 8 * (lane >> 4);
  v8h lo = *(const v8h*)p;
  v8h hi = *(const v8h*)(p + 16);
  return __builtin_shufflevector(lo, hi, 0, 1, 2, 3, 4, 5, 6, 7,
                                 8, 9, 10, 11, 12, 13, 14, 15);
}

// ---------------------------------------------------------------------------
// Pre-kernel 1: convert an f32 weight matrix to f16 (grid-stride)
// ---------------------------------------------------------------------------
__global__ __launch_bounds__(256) void cvt_f16(const float* __restrict__ src,
                                               _Float16* __restrict__ dst, int n) {
  for (int i = blockIdx.x * 256 + threadIdx.x; i < n; i += gridDim.x * 256)
    dst[i] = (_Float16)src[i];
}

// ---------------------------------------------------------------------------
// Pre-kernel 2: CPB MLP -> rpb[6][64][64] = 16*sigmoid(mlp(table)[rel_index])
// ---------------------------------------------------------------------------
__global__ __launch_bounds__(256) void cpb_kernel(
    const float* __restrict__ table, const int* __restrict__ rel_index,
    const float* __restrict__ w_cpb1, const float* __restrict__ b_cpb1,
    const float* __restrict__ w_cpb2, float* __restrict__ rpb) {
  __shared__ float bt[225][NH];
  const int tid = threadIdx.x;
  if (tid < 225) {
    float t0 = table[tid * 2 + 0], t1 = table[tid * 2 + 1];
    float acc[NH] = {0.f, 0.f, 0.f, 0.f, 0.f, 0.f};
    for (int j = 0; j < 512; ++j) {
      float hv = fmaxf(t0 * w_cpb1[j * 2] + t1 * w_cpb1[j * 2 + 1] + b_cpb1[j], 0.f);
#pragma unroll
      for (int h = 0; h < NH; ++h) acc[h] += w_cpb2[h * 512 + j] * hv;
    }
#pragma unroll
    for (int h = 0; h < NH; ++h) bt[tid][h] = acc[h];
  }
  __syncthreads();
  for (int idx = tid; idx < NTOK * NTOK; idx += 256) {
    int ri = rel_index[idx];
#pragma unroll
    for (int h = 0; h < NH; ++h) {
      float v = bt[ri][h];
      rpb[h * NTOK * NTOK + idx] = 16.f / (1.f + __expf(-v));
    }
  }
}

// ---------------------------------------------------------------------------
// Main fused window-attention kernel: one block (8 waves, wave32) per window
// ---------------------------------------------------------------------------
__global__ __launch_bounds__(256) void swinv2_window_attn(
    const float* __restrict__ x, const float* __restrict__ mask,
    const float* __restrict__ cat_v, const float* __restrict__ logit_scale,
    const _Float16* __restrict__ wqk_h, const float* __restrict__ q_bias,
    const float* __restrict__ k_bias, const _Float16* __restrict__ wv_h,
    const float* __restrict__ v_bias, const _Float16* __restrict__ wcatv_h,
    const float* __restrict__ v_mix_bias, const _Float16* __restrict__ wproj_h,
    const float* __restrict__ b_proj, const float* __restrict__ rpb,
    float* __restrict__ out) {
  __shared__ _Float16 s_x[NTOK * DIM];    // x (f16), re-staged with cat_v later
  __shared__ _Float16 s_q[NTOK * DIM];    // q (normalized); per-head O overwrites
  __shared__ _Float16 s_k[NTOK * DIM];    // k (normalized)
  __shared__ _Float16 s_vmT[DIM * NTOK];  // mixed v, TRANSPOSED: [dim][token]
  __shared__ __align__(16) unsigned char s_pool[NTOK * DIM * 2];  // v -> S^T(f32)+P(f16)
  __shared__ float s_scale[8];

  _Float16* s_v = (_Float16*)s_pool;                     // 24 KB (phase 1)
  float*    s_S = (float*)s_pool;                        // 16 KB, S TRANSPOSED [n][m]
  _Float16* s_p = (_Float16*)(s_pool + NTOK * NTOK * 4); // 8 KB, P row-major [m][n]

  const int tid  = threadIdx.x;
  const int lane = tid & 31;
  const int wv   = tid >> 5;
  const int b    = blockIdx.x;
  const float* xg  = x + (size_t)b * NTOK * DIM;
  const float* cvg = cat_v + (size_t)b * NTOK * DIM;
  const float* mk  = mask + (size_t)(b & (NWMSK - 1)) * NTOK * NTOK;

  // L2-resident f16 weights: CDNA5 global_prefetch hint
  __builtin_prefetch(wqk_h, 0, 1);
  __builtin_prefetch(wcatv_h, 0, 1);
  __builtin_prefetch(wproj_h, 0, 1);

  // stage x -> f16 LDS (vectorized: b128 load, b64 store); per-head scale
  {
    const float4* g4 = (const float4*)xg;
    v4h* s4 = (v4h*)s_x;
    for (int i = tid; i < NTOK * DIM / 4; i += 256) {
      float4 t = g4[i];
      v4h hh = {(_Float16)t.x, (_Float16)t.y, (_Float16)t.z, (_Float16)t.w};
      s4[i] = hh;
    }
  }
  if (tid < NH) s_scale[tid] = __expf(fminf(logit_scale[tid], 4.6051702f)); // log(100)
  __syncthreads();

  // ---- QK GEMM: x[64,192] @ w_qk^T[192,384]  -> 96 tiles of 16x16 ----
  for (int t = wv; t < 96; t += 8) {
    int r0 = (t / 24) * 16;
    int c0 = (t % 24) * 16;
    v8f acc = {};
#pragma unroll
    for (int s = 0; s < 6; ++s) {
      v16h a  = frag16(s_x, DIM, r0, s * 32, lane);
      v16h bf = frag16(wqk_h, DIM, c0, s * 32, lane);
      acc = WMMA_F16(a, bf, acc);
    }
    const int isk = (c0 >= DIM);
    const int cc  = (c0 % DIM) + (lane & 15);
    const float bias = isk ? k_bias[cc] : q_bias[cc];
    _Float16* dst = isk ? s_k : s_q;
    const int hi = lane >> 4;
#pragma unroll
    for (int r = 0; r < 8; ++r)
      dst[(r0 + r + 8 * hi) * DIM + cc] = (_Float16)(acc[r] + bias);
  }

  // ---- V GEMM: x @ w_v^T -> s_v ----
  for (int t = wv; t < 48; t += 8) {
    int r0 = (t / 12) * 16;
    int c0 = (t % 12) * 16;
    v8f acc = {};
#pragma unroll
    for (int s = 0; s < 6; ++s) {
      v16h a  = frag16(s_x, DIM, r0, s * 32, lane);
      v16h bf = frag16(wv_h, DIM, c0, s * 32, lane);
      acc = WMMA_F16(a, bf, acc);
    }
    const int cc = c0 + (lane & 15);
    const float bias = v_bias[cc];
    const int hi = lane >> 4;
#pragma unroll
    for (int r = 0; r < 8; ++r)
      s_v[(r0 + r + 8 * hi) * DIM + cc] = (_Float16)(acc[r] + bias);
  }
  __syncthreads();

  // ---- l2-normalize q (fold per-head scale) and k; re-stage s_x <- cat_v ----
  {
    const float4* g4 = (const float4*)cvg;
    v4h* s4 = (v4h*)s_x;
    for (int i = tid; i < NTOK * DIM / 4; i += 256) {
      float4 t = g4[i];
      v4h hh = {(_Float16)t.x, (_Float16)t.y, (_Float16)t.z, (_Float16)t.w};
      s4[i] = hh;
    }
  }
  for (int t = tid; t < NTOK * NH; t += 256) {
    int row = t / NH, head = t % NH;
    int base = row * DIM + head * HD;
    float sq = 0.f, sk = 0.f;
#pragma unroll
    for (int d = 0; d < HD; ++d) {
      float qv = (float)s_q[base + d];
      float kv = (float)s_k[base + d];
      sq += qv * qv;
      sk += kv * kv;
    }
    float fq = s_scale[head] / fmaxf(sqrtf(sq), 1e-12f);
    float fk = 1.0f / fmaxf(sqrtf(sk), 1e-12f);
#pragma unroll
    for (int d = 0; d < HD; ++d) {
      s_q[base + d] = (_Float16)((float)s_q[base + d] * fq);
      s_k[base + d] = (_Float16)((float)s_k[base + d] * fk);
    }
  }
  __syncthreads();

  // ---- V-mix GEMM: [v | cat_v] (64x384) @ w_catv^T -> s_vmT (transposed) ----
  for (int t = wv; t < 48; t += 8) {
    int r0 = (t / 12) * 16;
    int c0 = (t % 12) * 16;
    v8f acc = {};
#pragma unroll
    for (int s = 0; s < 6; ++s) {            // K = 0..191 from v (LDS)
      v16h a  = frag16(s_v, DIM, r0, s * 32, lane);
      v16h bf = frag16(wcatv_h, 2 * DIM, c0, s * 32, lane);
      acc = WMMA_F16(a, bf, acc);
    }
#pragma unroll
    for (int s = 0; s < 6; ++s) {            // K = 192..383 from cat_v (LDS)
      v16h a  = frag16(s_x, DIM, r0, s * 32, lane);
      v16h bf = frag16(wcatv_h, 2 * DIM, c0, DIM + s * 32, lane);
      acc = WMMA_F16(a, bf, acc);
    }
    const int cc = c0 + (lane & 15);
    const float bias = v_mix_bias[cc];
    const int hi = lane >> 4;
    // 8 consecutive tokens at fixed channel -> single 16B store (transposed layout)
    v8h o;
#pragma unroll
    for (int r = 0; r < 8; ++r) o[r] = (_Float16)(acc[r] + bias);
    *(v8h*)&s_vmT[cc * NTOK + r0 + 8 * hi] = o;
  }
  __syncthreads();

  // ---- attention, head by head ----
  for (int h = 0; h < NH; ++h) {
    // S = qn @ kn^T  (16 tiles, one K=32 WMMA each) + rpb + mask; store S^T
    for (int t = wv; t < 16; t += 8) {
      int r0 = (t >> 2) * 16;
      int c0 = (t & 3) * 16;
      v16h a  = frag16(s_q, DIM, r0, h * HD, lane);
      v16h bf = frag16(s_k, DIM, c0, h * HD, lane);
      v8f acc = {};
      acc = WMMA_F16(a, bf, acc);
      const int n  = c0 + (lane & 15);
      const int hi = lane >> 4;
      float vals[8];
#pragma unroll
      for (int r = 0; r < 8; ++r) {
        int m = r0 + r + 8 * hi;
        vals[r] = acc[r] + rpb[(h * NTOK + m) * NTOK + n] + mk[m * NTOK + n];
      }
      float4 f0 = {vals[0], vals[1], vals[2], vals[3]};
      float4 f1 = {vals[4], vals[5], vals[6], vals[7]};
      *(float4*)&s_S[n * NTOK + r0 + 8 * hi]     = f0;
      *(float4*)&s_S[n * NTOK + r0 + 8 * hi + 4] = f1;
    }
    __syncthreads();

    // row softmax over S^T -> P (row-major f16)
    if (tid < NTOK) {
      float mx = -1e30f;
      for (int j = 0; j < NTOK; ++j) mx = fmaxf(mx, s_S[j * NTOK + tid]);
      float sum = 0.f;
      for (int j = 0; j < NTOK; ++j) sum += __expf(s_S[j * NTOK + tid] - mx);
      float inv = 1.0f / sum;
      for (int j = 0; j < NTOK; ++j)
        s_p[tid * NTOK + j] = (_Float16)(__expf(s_S[j * NTOK + tid] - mx) * inv);
    }
    __syncthreads();

    // O_h = P @ V_h  (8 tiles, one per wave); overwrite dead q columns of head h
    {
      int r0 = (wv >> 1) * 16;
      int c0 = (wv & 1) * 16;
      v8f acc = {};
#pragma unroll
      for (int s = 0; s < 2; ++s) {
        v16h a  = frag16(s_p, NTOK, r0, s * 32, lane);
        v16h bf = frag16(s_vmT, NTOK, h * HD + c0, s * 32, lane);  // V^T rows = dims
        acc = WMMA_F16(a, bf, acc);
      }
      const int n  = h * HD + c0 + (lane & 15);
      const int hi = lane >> 4;
#pragma unroll
      for (int r = 0; r < 8; ++r)
        s_q[(r0 + r + 8 * hi) * DIM + n] = (_Float16)acc[r];
    }
    __syncthreads();
  }

  // ---- proj: O @ w_proj^T + b_proj -> global ----
  float* og = out + (size_t)b * NTOK * DIM;
  for (int t = wv; t < 48; t += 8) {
    int r0 = (t / 12) * 16;
    int c0 = (t % 12) * 16;
    v8f acc = {};
#pragma unroll
    for (int s = 0; s < 6; ++s) {
      v16h a  = frag16(s_q, DIM, r0, s * 32, lane);
      v16h bf = frag16(wproj_h, DIM, c0, s * 32, lane);
      acc = WMMA_F16(a, bf, acc);
    }
    const int cc = c0 + (lane & 15);
    const float bias = b_proj[cc];
    const int hi = lane >> 4;
#pragma unroll
    for (int r = 0; r < 8; ++r)
      og[(r0 + r + 8 * hi) * DIM + cc] = acc[r] + bias;
  }
}

extern "C" void kernel_launch(void* const* d_in, const int* in_sizes, int n_in,
                              void* d_out, int out_size, void* d_ws, size_t ws_size,
                              hipStream_t stream) {
  const float* x    = (const float*)d_in[0];
  const float* mask = (const float*)d_in[1];
  const float* cv   = (const float*)d_in[2];
  const float* ls   = (const float*)d_in[3];
  const float* tab  = (const float*)d_in[4];
  const int*   ri   = (const int*)d_in[5];
  const float* wqk  = (const float*)d_in[6];
  const float* qb   = (const float*)d_in[7];
  const float* kb   = (const float*)d_in[8];
  const float* wvv  = (const float*)d_in[9];
  const float* vb   = (const float*)d_in[10];
  const float* wcv  = (const float*)d_in[11];
  const float* vmb  = (const float*)d_in[12];
  const float* w1   = (const float*)d_in[13];
  const float* b1   = (const float*)d_in[14];
  const float* w2   = (const float*)d_in[15];
  const float* wp   = (const float*)d_in[16];
  const float* bp   = (const float*)d_in[17];
  float* out = (float*)d_out;

  // workspace layout: rpb f32 [6*64*64] | f16 weights (qk | v | catv | proj)
  float* rpb = (float*)d_ws;
  _Float16* wqk_h   = (_Float16*)((char*)d_ws + NH * NTOK * NTOK * 4);  // 73728 h
  _Float16* wv_h    = wqk_h + 2 * DIM * DIM;                            // 36864 h
  _Float16* wcatv_h = wv_h + DIM * DIM;                                 // 73728 h
  _Float16* wproj_h = wcatv_h + 2 * DIM * DIM;                          // 36864 h

  cpb_kernel<<<1, 256, 0, stream>>>(tab, ri, w1, b1, w2, rpb);
  cvt_f16<<<72, 256, 0, stream>>>(wqk, wqk_h, 2 * DIM * DIM);
  cvt_f16<<<36, 256, 0, stream>>>(wvv, wv_h, DIM * DIM);
  cvt_f16<<<72, 256, 0, stream>>>(wcv, wcatv_h, 2 * DIM * DIM);
  cvt_f16<<<36, 256, 0, stream>>>(wp, wproj_h, DIM * DIM);

  swinv2_window_attn<<<4096, 256, 0, stream>>>(x, mask, cv, ls, wqk_h, qb, kb, wv_h,
                                               vb, wcatv_h, vmb, wproj_h, bp, rpb, out);
}
